// OnlyLiquidEEG_16716012716776
// MI455X (gfx1250) — matrix operand order repaired and checked by reading, not compiled, not run on hardware
//
#include <hip/hip_runtime.h>
#include <hip/hip_bf16.h>

typedef __attribute__((ext_vector_type(16))) _Float16 v16h;
typedef __attribute__((ext_vector_type(8)))  _Float16 v8h;
typedef __attribute__((ext_vector_type(4)))  _Float16 v4h;
typedef __attribute__((ext_vector_type(8)))  float    v8f;
typedef __attribute__((ext_vector_type(4)))  unsigned int u32x4;
typedef __attribute__((ext_vector_type(4)))  int      i32x4;
typedef __attribute__((ext_vector_type(8)))  int      i32x8;

#define T_STEPS 8192
#define IN_F    8
#define CH      32                 // timesteps per TDM-staged chunk
#define NCH     (T_STEPS / CH)     // 256 chunks

// ---- workspace layout (bytes) ----
// [0, 45056)      : 44 WMMA B-fragments, f16, 512 halves each (bias folded in as an
//                   extra K-row matched by a constant-1 column in z)
// [45056, 49152)  : stage[512*2] f32 (final h2 before log-softmax)
static constexpr int FRAG_HALVES   = 512;
static constexpr int N_FRAGS       = 44;     // L0: 24, L1: 16, L2: 4
static constexpr int WS_STAGE_OFF  = N_FRAGS * FRAG_HALVES * 2;  // 45056

struct Params {
  const float* w[3][4];    // [layer][ff1,ff2,ta,tb], shape (hid, d)
  const float* b[3][4];
  const float* mask[3];    // (hid, d)
  const float* x;          // (512, 1, 8192, 8)
  _Float16*    frags;      // ws
  float*       stage;      // ws
  float*       out;        // d_out, (512, 2) f32
};

#if __has_builtin(__builtin_amdgcn_tanhf)
__device__ __forceinline__ float fast_tanh(float x) { return __builtin_amdgcn_tanhf(x); }
#else
__device__ __forceinline__ float fast_tanh(float x) {
  float e = __expf(2.f * x);
  return 1.f - 2.f * __builtin_amdgcn_rcpf(e + 1.f);
}
#endif
__device__ __forceinline__ float fast_sigmoid(float x) {
  return __builtin_amdgcn_rcpf(1.f + __expf(-x));
}
// One wave per workgroup; HW keeps same-wave LDS ops in order, so a compiler-only
// fence is sufficient instead of s_barrier.
__device__ __forceinline__ void lds_fence() { asm volatile("" ::: "memory"); }

__device__ __forceinline__ void wait_tensorcnt0() {
#if __has_builtin(__builtin_amdgcn_s_wait_tensorcnt)
  __builtin_amdgcn_s_wait_tensorcnt(0);
#else
  asm volatile("s_wait_tensorcnt 0" ::: "memory");
#endif
}

// ---- TDM: DMA one chunk (16 rows x CH*8 f32, row stride T*8 elems) into LDS ----
// D# packing per CDNA5 ISA.
__device__ __forceinline__ void tdm_issue_chunk(const float* gbase, void* ldsdst) {
#if __has_builtin(__builtin_amdgcn_tensor_load_to_lds)
  const unsigned long long ga = (unsigned long long)(uintptr_t)gbase;
  const unsigned la = (unsigned)(uintptr_t)ldsdst;       // low 32 bits = LDS byte addr
  u32x4 g0;
  g0[0] = 1u;                                            // count=1, user descriptor
  g0[1] = la;                                            // lds_addr
  g0[2] = (unsigned)ga;                                  // global_addr[31:0]
  g0[3] = (unsigned)((ga >> 32) & 0x01FFFFFFu) | (2u << 30);  // global_addr[56:32] | type=2
  const unsigned td0 = CH * IN_F;       // tensor_dim0 = 256 elements
  const unsigned td1 = 16;              // tensor_dim1 = 16 rows
  const unsigned tl0 = CH * IN_F;       // tile_dim0
  const unsigned tl1 = 16;              // tile_dim1
  const unsigned long long st0 = (unsigned long long)T_STEPS * IN_F;  // dim0 stride (elems)
  i32x8 g1;
  g1[0] = (int)(2u << 16);                               // data_size=2 -> 4-byte elems
  g1[1] = (int)((td0 & 0xFFFFu) << 16);                  // tensor_dim0[15:0]
  g1[2] = (int)((td0 >> 16) | ((td1 & 0xFFFFu) << 16));  // td0[31:16] | td1[15:0]
  g1[3] = (int)((td1 >> 16) | (tl0 << 16));              // td1[31:16] | tile_dim0
  g1[4] = (int)(tl1);                                    // tile_dim1 | tile_dim2=0
  g1[5] = (int)(st0 & 0xFFFFFFFFu);                      // stride0[31:0]
  g1[6] = (int)(st0 >> 32);                              // stride0[47:32] | stride1[15:0]=0
  g1[7] = 0;                                             // stride1[47:16]=0
  i32x4 gz = {0, 0, 0, 0};
#if __clang_major__ >= 23
  i32x8 gz8 = {0, 0, 0, 0, 0, 0, 0, 0};
  __builtin_amdgcn_tensor_load_to_lds(g0, g1, gz, gz, gz8, 0);
#else
  __builtin_amdgcn_tensor_load_to_lds(g0, g1, gz, gz, 0);
#endif
#else
  (void)gbase; (void)ldsdst;   // fallback handled by caller
#endif
}

#if __has_builtin(__builtin_amdgcn_tensor_load_to_lds)
#define HAVE_TDM 1
#else
#define HAVE_TDM 0
#endif

// ---------------- prep: pack masked weights (+bias K-row) into per-lane B-fragments ----------------
// B fragment (K=32 x N=16): lane L holds column n = L&15; halves j=0..15 hold
// K = (L<16 ? 0 : 16) + j. Contiguous 32B per lane -> one v16h register load.
__global__ void cfc_prep(Params p) {
  const int gid = blockIdx.x * blockDim.x + threadIdx.x;
  const int totalF = N_FRAGS * FRAG_HALVES; // 22528
  const int HID[3] = {38, 24, 2};
  const int DD[3]  = {46, 62, 26};
  if (gid >= totalF) return;
  int frag = gid >> 9;
  int h    = gid & 511;
  int lane = h >> 4;
  int j    = h & 15;
  int n    = lane & 15;
  int k    = ((lane >> 4) << 4) + j;
  int l, m, nt, kc;
  if (frag < 24)      { l = 0; int r = frag;      kc = r & 1; r >>= 1; nt = r % 3; m = r / 3; }
  else if (frag < 40) { l = 1; int r = frag - 24; kc = r & 1; r >>= 1; nt = r & 1; m = r >> 1; }
  else                { l = 2; m = frag - 40; nt = 0; kc = 0; }
  int row = nt * 16 + n;     // hidden unit
  int col = kc * 32 + k;     // input feature of z
  float v = 0.f;
  if (row < HID[l]) {
    if (col < DD[l]) {
      v = p.w[l][m][row * DD[l] + col];
      if (m < 2) v *= p.mask[l][row * DD[l] + col];   // NCP sparsity folded in
    } else if (col == DD[l]) {
      v = p.b[l][m][row];                             // bias row (z has a 1.0 there)
    }
  }
  p.frags[gid] = (_Float16)v;
}

// ---------------- main: one wave32 per 16-row batch tile, full T recurrence ----------------
__global__ __launch_bounds__(32) void cfc_main(Params p) {
  __shared__ __align__(16) _Float16 z0[16 * 64];  // [x(8) | h0(38) | 1@46 | pad]
  __shared__ __align__(16) _Float16 z1[16 * 64];  // [h0new(38) | h1(24) | 1@62 | pad]
  __shared__ __align__(16) _Float16 z2[16 * 32];  // [h1new(24) | h2(2) | 1@26 | pad]
  __shared__ __align__(16) float xstage[2][16 * CH * IN_F];  // double-buffered x chunks (32 KB)

  const int lane = threadIdx.x;
  const int tileBase = blockIdx.x * 16;
  const float* xtile = p.x + (size_t)tileBase * T_STEPS * IN_F;

  // kick off the first x chunk DMA before anything else
#if HAVE_TDM
  tdm_issue_chunk(xtile, &xstage[0][0]);
#endif

  // ---- preload all 44 B fragments into registers (352 VGPRs; wave32 has 1024) ----
  v16h bw[N_FRAGS];
#pragma unroll
  for (int f = 0; f < N_FRAGS; ++f)
    bw[f] = *reinterpret_cast<const v16h*>(p.frags + f * FRAG_HALVES + lane * 16);

  // h(0) = 0; padding columns stay zero forever; bias columns get 1.0
  for (int i = lane; i < 16 * 64; i += 32) { z0[i] = (_Float16)0; z1[i] = (_Float16)0; }
  for (int i = lane; i < 16 * 32; i += 32) z2[i] = (_Float16)0;
  if (lane < 16) {
    z0[lane * 64 + 46] = (_Float16)1.f;
    z1[lane * 64 + 62] = (_Float16)1.f;
    z2[lane * 32 + 26] = (_Float16)1.f;
  }
  lds_fence();

  constexpr int HID[3]      = {38, 24, 2};
  constexpr int NT[3]       = {3, 2, 1};
  constexpr int KC[3]       = {2, 2, 1};
  constexpr int ZW[3]       = {64, 64, 32};
  constexpr int SELF_OFF[3] = {8, 38, 24};
  _Float16* const zb[3] = {z0, z1, z2};

  const int xrow = lane >> 1;           // batch row in tile
  const int xc4  = (lane & 1) * 4;      // which half of the 8 features

#if !HAVE_TDM
  // fallback: stage chunk 0 with plain loads
  for (int i = lane; i < 16 * CH * IN_F / 4; i += 32) {
    int row = i >> 6, o = i & 63;
    float4 v = *reinterpret_cast<const float4*>(xtile + (size_t)row * T_STEPS * IN_F + o * 4);
    *reinterpret_cast<float4*>(&xstage[0][row * CH * IN_F + o * 4]) = v;
  }
#else
  wait_tensorcnt0();
#endif
  lds_fence();

  for (int c = 0; c < NCH; ++c) {
    const int buf = c & 1;
    // prefetch next chunk into the other buffer while we compute 32 steps
    if (c + 1 < NCH) {
#if HAVE_TDM
      tdm_issue_chunk(xtile + (size_t)(c + 1) * CH * IN_F, &xstage[buf ^ 1][0]);
#else
      for (int i = lane; i < 16 * CH * IN_F / 4; i += 32) {
        int row = i >> 6, o = i & 63;
        float4 v = *reinterpret_cast<const float4*>(
            xtile + (size_t)row * T_STEPS * IN_F + (size_t)(c + 1) * CH * IN_F + o * 4);
        *reinterpret_cast<float4*>(&xstage[buf ^ 1][row * CH * IN_F + o * 4]) = v;
      }
#endif
    }

#pragma unroll 1
    for (int s = 0; s < CH; ++s) {
      // ---- stage x_t into z0 cols 0..7 (f32 from LDS chunk -> f16) ----
      float4 xv = *reinterpret_cast<const float4*>(
          &xstage[buf][xrow * CH * IN_F + s * IN_F + xc4]);
      v4h xh; xh[0] = (_Float16)xv.x; xh[1] = (_Float16)xv.y;
              xh[2] = (_Float16)xv.z; xh[3] = (_Float16)xv.w;
      *reinterpret_cast<v4h*>(&z0[xrow * 64 + xc4]) = xh;
      lds_fence();

#pragma unroll
      for (int l = 0; l < 3; ++l) {
        _Float16* z = zb[l];
        const int zw = ZW[l];

        // ---- A fragments from z (16-bit A 16x32 layout, ISA 7.12.2) ----
        v16h a[2];
#pragma unroll
        for (int kc = 0; kc < KC[l]; ++kc) {
          const int m16 = lane & 15;
          const int hi  = lane >> 4;
          const _Float16* base = &z[m16 * zw + kc * 32 + hi * 16];
          v8h lo = *reinterpret_cast<const v8h*>(base);
          v8h hh = *reinterpret_cast<const v8h*>(base + 8);
          v16h av;
#pragma unroll
          for (int j = 0; j < 8; ++j) { av[j] = lo[j]; av[j + 8] = hh[j]; }
          a[kc] = av;
        }

#pragma unroll
        for (int nt = 0; nt < NT[l]; ++nt) {
          v8f acc[4];
#pragma unroll
          for (int m = 0; m < 4; ++m) {
            v8f cacc = {};   // first WMMA uses inline-0 SRC2; bias arrives via the 1-column
#pragma unroll
            for (int kc = 0; kc < KC[l]; ++kc) {
              const int frag = (l == 0) ? ((m * 3 + nt) * 2 + kc)
                             : (l == 1) ? (24 + (m * 2 + nt) * 2 + kc)
                                        : (40 + m);
              cacc = __builtin_amdgcn_wmma_f32_16x16x32_f16(false, a[kc], false, bw[frag],
                                                            (short)0, cacc, false, false);
            }
            acc[m] = cacc;
          }

          // ---- CfC gate + scatter h_new (D layout: row = r + 8*(lane>=16), col = lane&15) ----
          const int n     = lane & 15;
          const int gn    = nt * 16 + n;
          const bool valid = gn < HID[l];
          const int mbase = (lane >> 4) * 8;
#pragma unroll
          for (int r = 0; r < 8; ++r) {
            float f1 = fast_tanh(acc[0][r]);
            float f2 = fast_tanh(acc[1][r]);
            float ti = fast_sigmoid(acc[2][r] + acc[3][r]);
            float hn = f1 + ti * (f2 - f1);
            if (valid) {
              _Float16 h16 = (_Float16)hn;
              const int row = mbase + r;
              z[row * zw + SELF_OFF[l] + gn] = h16;              // recurrent state
              if (l < 2) zb[l + 1][row * ZW[l + 1] + gn] = h16;  // feed next layer
            }
          }
        }
        lds_fence();
      }
    }

    if (c + 1 < NCH) {
#if HAVE_TDM
      wait_tensorcnt0();     // next chunk's DMA had 32 steps to complete
#endif
      lds_fence();
    }
  }

  // final h2 (cols 24..25 of z2) -> stage
  {
    const int row = lane & 15;
    const int col = lane >> 4;
    p.stage[(tileBase + row) * 2 + col] = (float)z2[row * 32 + 24 + col];
  }
}

// ---------------- log_softmax over the batch axis (dim=0), per column ----------------
__global__ void cfc_logsoftmax(Params p) {
  __shared__ float red[512];
  const int b = threadIdx.x;
  float v[2] = {p.stage[b * 2 + 0], p.stage[b * 2 + 1]};
#pragma unroll
  for (int c = 0; c < 2; ++c) {
    red[b] = v[c];
    __syncthreads();
    for (int s = 256; s > 0; s >>= 1) {
      if (b < s) red[b] = fmaxf(red[b], red[b + s]);
      __syncthreads();
    }
    const float mx = red[0];
    __syncthreads();
    red[b] = __expf(v[c] - mx);
    __syncthreads();
    for (int s = 256; s > 0; s >>= 1) {
      if (b < s) red[b] += red[b + s];
      __syncthreads();
    }
    const float sm = red[0];
    __syncthreads();
    p.out[b * 2 + c] = v[c] - mx - __logf(sm);
  }
}

extern "C" void kernel_launch(void* const* d_in, const int* in_sizes, int n_in,
                              void* d_out, int out_size, void* d_ws, size_t ws_size,
                              hipStream_t stream) {
  (void)in_sizes; (void)n_in; (void)out_size; (void)ws_size;
  Params P;
  P.x = (const float*)d_in[0];
  for (int l = 0; l < 3; ++l) {
    const int base = 1 + l * 9;
    P.w[l][0] = (const float*)d_in[base + 0]; P.b[l][0] = (const float*)d_in[base + 1]; // ff1
    P.w[l][1] = (const float*)d_in[base + 2]; P.b[l][1] = (const float*)d_in[base + 3]; // ff2
    P.w[l][2] = (const float*)d_in[base + 4]; P.b[l][2] = (const float*)d_in[base + 5]; // ta
    P.w[l][3] = (const float*)d_in[base + 6]; P.b[l][3] = (const float*)d_in[base + 7]; // tb
    P.mask[l] = (const float*)d_in[base + 8];
  }
  P.frags = (_Float16*)d_ws;
  P.stage = (float*)((char*)d_ws + WS_STAGE_OFF);
  P.out   = (float*)d_out;

  cfc_prep<<<88, 256, 0, stream>>>(P);          // 22528 fragment halves (bias folded)
  cfc_main<<<32, 32, 0, stream>>>(P);           // 32 batch tiles x 1 wave32 each
  cfc_logsoftmax<<<1, 512, 0, stream>>>(P);     // cross-batch reduction
}
